// seg_model_18614388261212
// MI455X (gfx1250) — matrix operand order repaired and checked by reading, not compiled, not run on hardware
//
#include <hip/hip_runtime.h>
#include <hip/hip_bf16.h>
#include <cstdint>

// ---------------------------------------------------------------------------
// CDNA5 / gfx1250 WMMA helpers (wave32)
// ---------------------------------------------------------------------------
typedef __attribute__((ext_vector_type(16))) __bf16 v16bf;
typedef __attribute__((ext_vector_type(8)))  __bf16 v8bf;
typedef __attribute__((ext_vector_type(8)))  float  v8f;

__device__ __forceinline__ v8f wmma_bf16(v16bf a, v16bf b, v8f c) {
  // D = A(16x32 bf16) * B(32x16 bf16) + C(16x16 f32)
  return __builtin_amdgcn_wmma_f32_16x16x32_bf16(
      /*neg_a=*/false, a, /*neg_b=*/false, b,
      /*c_mod=*/(short)0, c, /*reuse_a=*/false, /*reuse_b=*/false);
}

// A fragment for 16x16x32 bf16 (per ISA 7.12.2): lane l, row m=l&15,
// off = (l>=16)?8:0 ; elements = row[off..off+7] ++ row[16+off..16+off+7]
__device__ __forceinline__ v16bf load_afrag(const __bf16* p) {
  v8bf a0 = *(const v8bf*)(p);
  v8bf a1 = *(const v8bf*)(p + 16);
  return __builtin_shufflevector(a0, a1, 0,1,2,3,4,5,6,7,8,9,10,11,12,13,14,15);
}

__device__ __forceinline__ __bf16 f2bf(float f) { return (__bf16)f; }
__device__ __forceinline__ float  bf2f(__bf16 h) { return (float)h; }

#define NPTS 10000
#define CDIM 1024
#define NTIL 625      // NPTS / 16

// ---------------------------------------------------------------------------
// top-k (k=4, LARGEST value, ties -> smaller index first; matches jax top_k)
// ---------------------------------------------------------------------------
__device__ __forceinline__ bool tk_better(float nv, int ni, float ov, int oi) {
  return (nv > ov) || ((nv == ov) && (ni < oi));
}
__device__ __forceinline__ void topk_insert(float (&v)[4], int (&id)[4],
                                            float nv, int ni) {
  if (!tk_better(nv, ni, v[3], id[3])) return;
  v[3] = nv; id[3] = ni;
#pragma unroll
  for (int s = 3; s >= 1; --s) {
    if (tk_better(v[s], id[s], v[s-1], id[s-1])) {
      float tf = v[s]; v[s] = v[s-1]; v[s-1] = tf;
      int   ti = id[s]; id[s] = id[s-1]; id[s-1] = ti;
    }
  }
}

// ---------------------------------------------------------------------------
// Weight pre-swizzle into B-fragment order:
//   out[((ct*KC + kc)*32 + lane)*16 + e] = W[rowOff + k][n],  zero-pad k>=Kreal
//   n = ct*16 + (lane&15), k = kc*32 + (lane>>4)*8 + (e&7) + (e>>3)*16
// ---------------------------------------------------------------------------
__global__ void swizzleB_kernel(const float* __restrict__ W, int ldn, int rowOff,
                                int Kreal, int KC, int total,
                                __bf16* __restrict__ hi, __bf16* __restrict__ lo) {
  int tid = blockIdx.x * blockDim.x + threadIdx.x;
  if (tid >= total) return;
  int e    = tid & 15;
  int lane = (tid >> 4) & 31;
  int rest = tid >> 9;
  int kc   = rest % KC;
  int ct   = rest / KC;
  int n = ct * 16 + (lane & 15);
  int k = kc * 32 + ((lane >> 4) * 8) + (e & 7) + ((e >> 3) * 16);
  float w = (k < Kreal) ? W[(size_t)(rowOff + k) * ldn + n] : 0.0f;
  __bf16 h = f2bf(w);
  hi[tid] = h;
  lo[tid] = f2bf(w - bf2f(h));
}

// ---------------------------------------------------------------------------
// Per-point MLP: feature = relu(relu(p@W1+b1)@W2+b2)   (N x 64, also bf16 hi/lo)
// block = 256 threads = 4 points x 64 channels
// ---------------------------------------------------------------------------
__global__ __launch_bounds__(256) void mlp12_kernel(
    const float* __restrict__ pts,
    const float* __restrict__ W1, const float* __restrict__ b1,
    const float* __restrict__ W2, const float* __restrict__ b2,
    float* __restrict__ feat, __bf16* __restrict__ fhi, __bf16* __restrict__ flo) {
  __shared__ float h1s[4][64];
  int tid = threadIdx.x;
  int g = tid >> 6, j = tid & 63;
  int p = blockIdx.x * 4 + g;
  float px = pts[p*3+0], py = pts[p*3+1], pz = pts[p*3+2];
  float h1 = b1[j] + px*W1[0*64+j] + py*W1[1*64+j] + pz*W1[2*64+j];
  h1s[g][j] = fmaxf(h1, 0.0f);
  __syncthreads();
  float f = b2[j];
#pragma unroll 8
  for (int i = 0; i < 64; ++i) f += h1s[g][i] * W2[i*64+j];
  f = fmaxf(f, 0.0f);
  size_t o = (size_t)p*64 + j;
  feat[o] = f;
  __bf16 h = f2bf(f);
  fhi[o] = h; flo[o] = f2bf(f - bf2f(h));
}

// ---------------------------------------------------------------------------
// feature2 = relu(feature @ W3 + b3) : (N x 64)@(64 x 1024)
// one wave per 16x16 output tile; jobs = 625 rowtiles * 64 coltiles
// ---------------------------------------------------------------------------
__global__ __launch_bounds__(256) void feat2_kernel(
    const __bf16* __restrict__ fhi, const __bf16* __restrict__ flo,
    const __bf16* __restrict__ w3hi, const __bf16* __restrict__ w3lo,
    const float* __restrict__ b3,
    float* __restrict__ f2, __bf16* __restrict__ xhi, __bf16* __restrict__ xlo) {
  int wave = threadIdx.x >> 5, lane = threadIdx.x & 31;
  int job = blockIdx.x * 8 + wave;            // < 40000
  int rt = job >> 6, ct = job & 63;
  int m = lane & 15, hlf = lane >> 4;
  const __bf16* arh = fhi + (size_t)(rt*16 + m)*64 + hlf*8;
  const __bf16* arl = flo + (size_t)(rt*16 + m)*64 + hlf*8;
  v8f acc = {};
#pragma unroll
  for (int kc = 0; kc < 2; ++kc) {
    v16bf ah = load_afrag(arh + kc*32);
    v16bf al = load_afrag(arl + kc*32);
    size_t bo = (size_t)((ct*2 + kc)*32 + lane) * 16;
    v16bf bh = *(const v16bf*)(w3hi + bo);
    v16bf bl = *(const v16bf*)(w3lo + bo);
    acc = wmma_bf16(ah, bh, acc);
    acc = wmma_bf16(ah, bl, acc);
    acc = wmma_bf16(al, bh, acc);
  }
  int col = ct*16 + m;
  float bias = b3[col];
#pragma unroll
  for (int r = 0; r < 8; ++r) {
    int row = rt*16 + r + 8*hlf;
    float v = fmaxf(acc[r] + bias, 0.0f);
    size_t o = (size_t)row * CDIM + col;
    f2[o] = v;
    __bf16 h = f2bf(v);
    xhi[o] = h; xlo[o] = f2bf(v - bf2f(h));
  }
}

// ---------------------------------------------------------------------------
// xx[i] = sum_c f2[i][c]^2    (one wave per row)
// ---------------------------------------------------------------------------
__global__ __launch_bounds__(256) void rownorm_kernel(const float* __restrict__ f2,
                                                      float* __restrict__ xx) {
  int row = blockIdx.x * 8 + (threadIdx.x >> 5);
  int lane = threadIdx.x & 31;
  const float* r = f2 + (size_t)row * CDIM;
  float s = 0.0f;
  for (int t = lane; t < CDIM; t += 32) { float x = r[t]; s += x * x; }
#pragma unroll
  for (int off = 16; off > 0; off >>= 1) s += __shfl_xor(s, off, 32);
  if (lane == 0) xx[row] = s;
}

// gmax[c] = max_i f2[i][c]
__global__ void gmax_kernel(const float* __restrict__ f2, float* __restrict__ gmax) {
  int c = blockIdx.x * blockDim.x + threadIdx.x;   // 1024 total
  float m = -__FLT_MAX__;
  for (int i = 0; i < NPTS; ++i) m = fmaxf(m, f2[(size_t)i * CDIM + c]);
  gmax[c] = m;
}

// gvec[j] = b4[j] + sum_c gmax[c]*W4[c][j]   (folds the broadcast gmax block)
__global__ void gvec_kernel(const float* __restrict__ gmax,
                            const float* __restrict__ W4, const float* __restrict__ b4,
                            float* __restrict__ gvec) {
  int j = threadIdx.x;   // 512
  float s = b4[j];
  for (int c = 0; c < CDIM; ++c) s += gmax[c] * W4[(size_t)c * 512 + j];
  gvec[j] = s;
}

// ---------------------------------------------------------------------------
// Fused gram + top-4: per block, one 16-row tile vs all 10000 columns.
// sq(i,j) = xx_i + xx_j - 2*dot(i,j), dot via split-bf16 WMMA (3/K-chunk).
// Keep top-4 LARGEST sq per row (reference drops rank 0, emits -sq ranks 1..3).
// ---------------------------------------------------------------------------
struct Cand { float v; int i; };
union GramSM {
  __bf16 a[2][16][CDIM];     // 64 KB staged A tile (hi/lo)
  Cand   cand[16][512];      // 64 KB merge scratch (reused after barrier)
};

__global__ __launch_bounds__(256) void gram_topk_kernel(
    const __bf16* __restrict__ xhi, const __bf16* __restrict__ xlo,
    const float* __restrict__ xx,
    int* __restrict__ knn_idx, float* __restrict__ knn_dist) {
  __shared__ __align__(16) GramSM sm;
  int tid = threadIdx.x;
  int i0 = blockIdx.x * 16;

  // stage 16x1024 hi/lo A rows (contiguous region) into LDS
  {
    const uint4* sh = (const uint4*)(xhi + (size_t)i0 * CDIM);
    const uint4* sl = (const uint4*)(xlo + (size_t)i0 * CDIM);
    uint4* dh = (uint4*)(&sm.a[0][0][0]);
    uint4* dl = (uint4*)(&sm.a[1][0][0]);
    for (int t = tid; t < 2048; t += 256) { dh[t] = sh[t]; dl[t] = sl[t]; }
  }
  __syncthreads();

  int wave = tid >> 5, lane = tid & 31;
  int m = lane & 15, hlf = lane >> 4;

  float xxi[8];
#pragma unroll
  for (int r = 0; r < 8; ++r) xxi[r] = xx[i0 + r + 8*hlf];

  float tv[8][4]; int ti[8][4];
#pragma unroll
  for (int r = 0; r < 8; ++r)
#pragma unroll
    for (int s = 0; s < 4; ++s) { tv[r][s] = -__FLT_MAX__; ti[r][s] = 0x7fffffff; }

  const __bf16* arh = &sm.a[0][m][hlf*8];
  const __bf16* arl = &sm.a[1][m][hlf*8];

  for (int jt = wave; jt < NTIL; jt += 8) {
    int j0 = jt * 16;
    const __bf16* brh = xhi + (size_t)(j0 + m) * CDIM + hlf*8;
    const __bf16* brl = xlo + (size_t)(j0 + m) * CDIM + hlf*8;
    v8f acc = {};
#pragma unroll 4
    for (int kc = 0; kc < 32; ++kc) {
      int kb = kc * 32;
      v16bf ah = load_afrag(arh + kb);
      v16bf al = load_afrag(arl + kb);
      v16bf bh = load_afrag(brh + kb);
      v16bf bl = load_afrag(brl + kb);
      acc = wmma_bf16(ah, bh, acc);
      acc = wmma_bf16(ah, bl, acc);
      acc = wmma_bf16(al, bh, acc);
    }
    float xxj = xx[j0 + m];
    int colg = j0 + m;
#pragma unroll
    for (int r = 0; r < 8; ++r) {
      float sq = xxi[r] + xxj - 2.0f * acc[r];
      topk_insert(tv[r], ti[r], sq, colg);
    }
  }
  __syncthreads();   // A staging no longer needed; reuse LDS as merge scratch

  // dump per-lane candidate lists: per row, 8 waves * 16 lanes = 128 lists of 4
#pragma unroll
  for (int r = 0; r < 8; ++r) {
    int row = r + 8*hlf;
    int base = (wave * 16 + m) * 4;
#pragma unroll
    for (int s = 0; s < 4; ++s) {
      sm.cand[row][base + s].v = tv[r][s];
      sm.cand[row][base + s].i = ti[r][s];
    }
  }
  __syncthreads();

  if (tid < 16) {
    float bv[4]; int bi[4];
#pragma unroll
    for (int s = 0; s < 4; ++s) { bv[s] = -__FLT_MAX__; bi[s] = 0x7fffffff; }
    for (int c = 0; c < 512; ++c) {
      Cand cd = sm.cand[tid][c];
      topk_insert(bv, bi, cd.v, cd.i);
    }
    int row = i0 + tid;
#pragma unroll
    for (int s = 1; s < 4; ++s) {
      knn_idx[row*3 + (s-1)]  = bi[s];
      knn_dist[row*3 + (s-1)] = -bv[s];   // reference emits pd = -sq
    }
  }
}

// ---------------------------------------------------------------------------
// Final fused MLP: h4 = relu(gvec + tail@W4tail); h5 = relu(h4@W5+b5);
// out = softmax(h5@W6+b6). One block per 16 points.
// ---------------------------------------------------------------------------
__global__ __launch_bounds__(256) void final_mlp_kernel(
    const float* __restrict__ feat, const float* __restrict__ pts,
    const int* __restrict__ knn_idx, const float* __restrict__ knn_dist,
    const __bf16* __restrict__ w4hi, const __bf16* __restrict__ w4lo,
    const float* __restrict__ gvec,
    const __bf16* __restrict__ w5hi, const __bf16* __restrict__ w5lo,
    const float* __restrict__ b5,
    const float* __restrict__ W6, const float* __restrict__ b6,
    float* __restrict__ out) {
  __shared__ __align__(16) __bf16 ta_hi[16][96], ta_lo[16][96];
  __shared__ __align__(16) __bf16 h4hi[16][512], h4lo[16][512];
  __shared__ __align__(16) float  h5[16][256];
  int tid = threadIdx.x;
  int i0 = blockIdx.x * 16;

  // build tail A (feature 0..63 | knn 64..75 | zero pad 76..95), bf16 hi/lo
  for (int t = tid; t < 16*96; t += 256) {
    int r = t / 96, c = t % 96;
    int row = i0 + r;
    float v = 0.0f;
    if (c < 64) {
      v = feat[(size_t)row*64 + c];
    } else if (c < 76) {
      int cc = c - 64, k = cc >> 2, comp = cc & 3;
      if (comp == 3) v = knn_dist[row*3 + k];
      else {
        int nb = knn_idx[row*3 + k];
        v = pts[nb*3 + comp] - pts[row*3 + comp];
      }
    }
    __bf16 h = f2bf(v);
    ta_hi[r][c] = h; ta_lo[r][c] = f2bf(v - bf2f(h));
  }
  __syncthreads();

  int wave = tid >> 5, lane = tid & 31;
  int m = lane & 15, hlf = lane >> 4;

  // GEMM1: (16 x 96) @ (96 x 512), K padded; epilogue adds gvec, relu
  for (int ct = wave; ct < 32; ct += 8) {
    v8f acc = {};
#pragma unroll
    for (int kc = 0; kc < 3; ++kc) {
      v16bf ah = load_afrag(&ta_hi[m][kc*32 + hlf*8]);
      v16bf al = load_afrag(&ta_lo[m][kc*32 + hlf*8]);
      size_t bo = (size_t)((ct*3 + kc)*32 + lane) * 16;
      v16bf bh = *(const v16bf*)(w4hi + bo);
      v16bf bl = *(const v16bf*)(w4lo + bo);
      acc = wmma_bf16(ah, bh, acc);
      acc = wmma_bf16(ah, bl, acc);
      acc = wmma_bf16(al, bh, acc);
    }
    int col = ct*16 + m;
    float gv = gvec[col];
#pragma unroll
    for (int r = 0; r < 8; ++r) {
      int row = r + 8*hlf;
      float h = fmaxf(acc[r] + gv, 0.0f);
      __bf16 hb = f2bf(h);
      h4hi[row][col] = hb; h4lo[row][col] = f2bf(h - bf2f(hb));
    }
  }
  __syncthreads();

  // GEMM2: (16 x 512) @ (512 x 256) + b5, relu
  for (int ct = wave; ct < 16; ct += 8) {
    v8f acc = {};
#pragma unroll 4
    for (int kc = 0; kc < 16; ++kc) {
      v16bf ah = load_afrag(&h4hi[m][kc*32 + hlf*8]);
      v16bf al = load_afrag(&h4lo[m][kc*32 + hlf*8]);
      size_t bo = (size_t)((ct*16 + kc)*32 + lane) * 16;
      v16bf bh = *(const v16bf*)(w5hi + bo);
      v16bf bl = *(const v16bf*)(w5lo + bo);
      acc = wmma_bf16(ah, bh, acc);
      acc = wmma_bf16(ah, bl, acc);
      acc = wmma_bf16(al, bh, acc);
    }
    int col = ct*16 + m;
    float bb = b5[col];
#pragma unroll
    for (int r = 0; r < 8; ++r)
      h5[r + 8*hlf][col] = fmaxf(acc[r] + bb, 0.0f);
  }
  __syncthreads();

  // head: (256 -> 6) + softmax, one thread per point
  if (tid < 16) {
    float z[6];
#pragma unroll
    for (int j = 0; j < 6; ++j) z[j] = b6[j];
    for (int c = 0; c < 256; ++c) {
      float hv = h5[tid][c];
#pragma unroll
      for (int j = 0; j < 6; ++j) z[j] += hv * W6[c*6 + j];
    }
    float mx = z[0];
#pragma unroll
    for (int j = 1; j < 6; ++j) mx = fmaxf(mx, z[j]);
    float s = 0.0f;
#pragma unroll
    for (int j = 0; j < 6; ++j) { z[j] = __expf(z[j] - mx); s += z[j]; }
    float inv = 1.0f / s;
    int row = i0 + tid;
#pragma unroll
    for (int j = 0; j < 6; ++j) out[row*6 + j] = z[j] * inv;
  }
}

// ---------------------------------------------------------------------------
// host launcher
// ---------------------------------------------------------------------------
extern "C" void kernel_launch(void* const* d_in, const int* in_sizes, int n_in,
                              void* d_out, int out_size, void* d_ws, size_t ws_size,
                              hipStream_t stream) {
  (void)in_sizes; (void)n_in; (void)out_size; (void)ws_size;
  const float* pts = (const float*)d_in[0];
  const float* W1 = (const float*)d_in[1];  const float* b1 = (const float*)d_in[2];
  const float* W2 = (const float*)d_in[3];  const float* b2 = (const float*)d_in[4];
  const float* W3 = (const float*)d_in[5];  const float* b3 = (const float*)d_in[6];
  const float* W4 = (const float*)d_in[7];  const float* b4 = (const float*)d_in[8];
  const float* W5 = (const float*)d_in[9];  const float* b5 = (const float*)d_in[10];
  const float* W6 = (const float*)d_in[11]; const float* b6 = (const float*)d_in[12];
  float* out = (float*)d_out;

  char* p = (char*)d_ws;
  auto alloc = [&](size_t n) -> void* {
    void* r = (void*)p; p += (n + 255) & ~(size_t)255; return r;
  };
  float*  feat  = (float*) alloc((size_t)NPTS * 64 * 4);
  __bf16* fhi   = (__bf16*)alloc((size_t)NPTS * 64 * 2);
  __bf16* flo   = (__bf16*)alloc((size_t)NPTS * 64 * 2);
  float*  f2    = (float*) alloc((size_t)NPTS * CDIM * 4);
  __bf16* xhi   = (__bf16*)alloc((size_t)NPTS * CDIM * 2);
  __bf16* xlo   = (__bf16*)alloc((size_t)NPTS * CDIM * 2);
  float*  xx    = (float*) alloc((size_t)NPTS * 4);
  float*  gmax  = (float*) alloc((size_t)CDIM * 4);
  float*  gvec  = (float*) alloc(512 * 4);
  int*    kidx  = (int*)   alloc((size_t)NPTS * 3 * 4);
  float*  kdst  = (float*) alloc((size_t)NPTS * 3 * 4);
  __bf16* w3hi  = (__bf16*)alloc((size_t)64  * CDIM * 2);
  __bf16* w3lo  = (__bf16*)alloc((size_t)64  * CDIM * 2);
  __bf16* w4thi = (__bf16*)alloc((size_t)96  * 512 * 2);
  __bf16* w4tlo = (__bf16*)alloc((size_t)96  * 512 * 2);
  __bf16* w5hi  = (__bf16*)alloc((size_t)512 * 256 * 2);
  __bf16* w5lo  = (__bf16*)alloc((size_t)512 * 256 * 2);

  // weight pre-swizzles into B-fragment order
  swizzleB_kernel<<<(64*CDIM + 255)/256, 256, 0, stream>>>(
      W3, 1024, 0, 64, 2, 64*CDIM, w3hi, w3lo);
  swizzleB_kernel<<<(96*512 + 255)/256, 256, 0, stream>>>(
      W4, 512, 1024, 76, 3, 96*512, w4thi, w4tlo);
  swizzleB_kernel<<<(512*256 + 255)/256, 256, 0, stream>>>(
      W5, 256, 0, 512, 16, 512*256, w5hi, w5lo);

  mlp12_kernel<<<NPTS/4, 256, 0, stream>>>(pts, W1, b1, W2, b2, feat, fhi, flo);
  feat2_kernel<<<(NTIL*64)/8, 256, 0, stream>>>(fhi, flo, w3hi, w3lo, b3, f2, xhi, xlo);
  rownorm_kernel<<<NPTS/8, 256, 0, stream>>>(f2, xx);
  gmax_kernel<<<CDIM/256, 256, 0, stream>>>(f2, gmax);
  gram_topk_kernel<<<NTIL, 256, 0, stream>>>(xhi, xlo, xx, kidx, kdst);
  gvec_kernel<<<1, 512, 0, stream>>>(gmax, W4, b4, gvec);
  final_mlp_kernel<<<NTIL, 256, 0, stream>>>(feat, pts, kidx, kdst,
                                             w4thi, w4tlo, gvec,
                                             w5hi, w5lo, b5, W6, b6, out);
}
